// MLPMoE_65738769433448
// MI455X (gfx1250) — compile-verified, matrix operand order checked by reading, not run on hardware
//
#include <hip/hip_runtime.h>
#include <hip/hip_bf16.h>
#include <math.h>

// Problem constants (from reference): E,K,D,H,B,T = 8,2,1024,4096,2,576
#define E_    8
#define TOPK_ 2
#define D_    1024
#define H_    4096
#define B_    2
#define T_    576
#define BT_   (B_ * T_)

#define LDA 40  // LDS row stride in bf16 elems: 80 B, multiple of 16 B for b128 reads

typedef __attribute__((ext_vector_type(16))) __bf16 v16bf;
typedef __attribute__((ext_vector_type(2)))  __bf16 v2bf;
typedef __attribute__((ext_vector_type(2)))  float  v2f;
typedef __attribute__((ext_vector_type(8)))  float  v8f;

union FragBF {
  v16bf v;
  uint4 q[2];
};

// native f32 -> bf16 (RNE) via hardware convert
static __device__ __forceinline__ unsigned short f2bf(float f) {
  __bf16 h = (__bf16)f;
  return __builtin_bit_cast(unsigned short, h);
}
// pack two converts into one dword: vector convert lowers to a single
// v_cvt_pk_bf16_f32 dst, lo, hi (no halfword regrouping)
static __device__ __forceinline__ unsigned pack2bf(float lo, float hi) {
  v2f f = {lo, hi};
  v2bf b = __builtin_convertvector(f, v2bf);
  return __builtin_bit_cast(unsigned, b);
}

// ---------------------------------------------------------------------------
// Kernel 1: convert x (fp32 [BT,D]) -> bf16 copy used as WMMA A operand
// ---------------------------------------------------------------------------
__global__ void cvt_x_kernel(const float* __restrict__ x,
                             unsigned* __restrict__ xb32) {
  int i = (blockIdx.x * blockDim.x + threadIdx.x) * 4;
  float4 f = *(const float4*)(x + i);
  uint2 o;
  o.x = pack2bf(f.x, f.y);
  o.y = pack2bf(f.z, f.w);
  *(uint2*)(xb32 + i / 2) = o;
}

// ---------------------------------------------------------------------------
// Kernel 2: gating. One wave32 per token: logits = x[t,:] @ gate_w, softmax,
// top-2, renormalize, atomic slot assignment into per-expert token lists.
// ---------------------------------------------------------------------------
__global__ void gate_kernel(const float* __restrict__ x,
                            const float* __restrict__ gw,
                            int* __restrict__ counts,
                            int* __restrict__ tok_list,    // [E][BT]
                            float* __restrict__ wgt_list,  // [E][BT]
                            int* __restrict__ tok_e,       // [BT][2]
                            int* __restrict__ tok_s) {     // [BT][2]
  const int wave = threadIdx.x >> 5;
  const int lane = threadIdx.x & 31;
  const int t = blockIdx.x * 8 + wave;
  if (t >= BT_) return;

  float acc[E_];
#pragma unroll
  for (int e = 0; e < E_; ++e) acc[e] = 0.0f;

  for (int d = lane; d < D_; d += 32) {
    float xv = x[(size_t)t * D_ + d];
    const float* g = gw + (size_t)d * E_;
#pragma unroll
    for (int e = 0; e < E_; ++e) acc[e] = fmaf(xv, g[e], acc[e]);
  }
#pragma unroll
  for (int off = 16; off > 0; off >>= 1) {
#pragma unroll
    for (int e = 0; e < E_; ++e) acc[e] += __shfl_down(acc[e], off, 32);
  }

  if (lane == 0) {
    float m = acc[0];
#pragma unroll
    for (int e = 1; e < E_; ++e) m = fmaxf(m, acc[e]);
    float p[E_];
#pragma unroll
    for (int e = 0; e < E_; ++e) p[e] = __expf(acc[e] - m);
    int i0 = 0;
#pragma unroll
    for (int e = 1; e < E_; ++e) if (p[e] > p[i0]) i0 = e;
    int i1 = (i0 == 0) ? 1 : 0;
#pragma unroll
    for (int e = 0; e < E_; ++e) if (e != i0 && p[e] > p[i1]) i1 = e;
    float ws = p[i0] + p[i1];
    float w0 = p[i0] / ws, w1 = p[i1] / ws;

    int s0 = atomicAdd(&counts[i0], 1);
    tok_list[i0 * BT_ + s0] = t;
    wgt_list[i0 * BT_ + s0] = w0;
    tok_e[2 * t + 0] = i0;
    tok_s[2 * t + 0] = s0;

    int s1 = atomicAdd(&counts[i1], 1);
    tok_list[i1 * BT_ + s1] = t;
    wgt_list[i1 * BT_ + s1] = w1;
    tok_e[2 * t + 1] = i1;
    tok_s[2 * t + 1] = s1;
  }
}

// ---------------------------------------------------------------------------
// Kernel 3: exclusive scan of per-expert counts -> compact row offsets
// ---------------------------------------------------------------------------
__global__ void scan_kernel(const int* __restrict__ counts,
                            int* __restrict__ offsets) {
  if (threadIdx.x == 0) {
    int o = 0;
    for (int e = 0; e < E_; ++e) { offsets[e] = o; o += counts[e]; }
    offsets[E_] = o;
  }
}

// ---------------------------------------------------------------------------
// GEMM1: h = gelu(gather(x)[rows,D] @ w1[e] + b1[e]) -> bf16 [rows,H]
// Block tile 64(M) x 128(N), 8 waves each computing a 32x32 sub-tile via four
// v_wmma_f32_16x16x32_bf16 accumulators. K slab = 32, double-buffered LDS:
// global loads of slab s+1 overlap the WMMA work on slab s. w tiles are read
// fp32 and converted to bf16 (v_cvt_pk_bf16_f32) while staged transposed.
// ---------------------------------------------------------------------------
__global__ __launch_bounds__(256)
void gemm1_kernel(const unsigned short* __restrict__ xb,
                  const float* __restrict__ w1,
                  const float* __restrict__ b1,
                  const int* __restrict__ counts,
                  const int* __restrict__ offsets,
                  const int* __restrict__ tok_list,
                  unsigned short* __restrict__ hbuf) {
  const int e = blockIdx.z, mt = blockIdx.y, nt = blockIdx.x;
  const int cnt = counts[e];
  if (mt * 64 >= cnt) return;
  const int base = offsets[e];

  __shared__ __align__(16) unsigned short As[2][64 * LDA];   // [m][k]
  __shared__ __align__(16) unsigned short Bs[2][128 * LDA];  // [n][k] transposed

  const int tid = threadIdx.x;
  const int wave = tid >> 5, lane = tid & 31;
  const int moff = (wave & 1) * 32, noff = (wave >> 1) * 32;
  const int lrow = lane & 15, khalf = lane >> 4;

  // A loader: 64 rows x 32 k = one b128 (8 bf16) per thread
  const int arow = tid >> 2;
  const int aseg = (tid & 3) * 8;
  const int gra = mt * 64 + arow;
  const int tok = (gra < cnt) ? tok_list[e * BT_ + gra] : -1;
  const unsigned short* asrc =
      (tok >= 0) ? (xb + (size_t)tok * D_ + aseg) : nullptr;

  // B loader: 128 n x 32 k fp32 = 16 floats per thread, k-pair per column so
  // two converts pack into one aligned b32 LDS store
  const int bn = (tid & 15) * 8;
  const int bkp = (tid >> 4) * 2;
  const float* wsrc = w1 + (size_t)e * D_ * H_ + (size_t)nt * 128 + bn;

  uint4 areg;
  float4 bA0, bA1, bB0, bB1;

  auto load_slab = [&](int k0) {
    areg = make_uint4(0u, 0u, 0u, 0u);
    if (asrc) areg = *(const uint4*)(asrc + k0);
    const float* p0 = wsrc + (size_t)(k0 + bkp) * H_;
    bA0 = *(const float4*)(p0);
    bA1 = *(const float4*)(p0 + 4);
    bB0 = *(const float4*)(p0 + H_);
    bB1 = *(const float4*)(p0 + H_ + 4);
  };
  auto store_slab = [&](int buf) {
    *(uint4*)(&As[buf][arow * LDA + aseg]) = areg;
    unsigned short* bs = &Bs[buf][0];
    *(unsigned*)(bs + (bn + 0) * LDA + bkp) = pack2bf(bA0.x, bB0.x);
    *(unsigned*)(bs + (bn + 1) * LDA + bkp) = pack2bf(bA0.y, bB0.y);
    *(unsigned*)(bs + (bn + 2) * LDA + bkp) = pack2bf(bA0.z, bB0.z);
    *(unsigned*)(bs + (bn + 3) * LDA + bkp) = pack2bf(bA0.w, bB0.w);
    *(unsigned*)(bs + (bn + 4) * LDA + bkp) = pack2bf(bA1.x, bB1.x);
    *(unsigned*)(bs + (bn + 5) * LDA + bkp) = pack2bf(bA1.y, bB1.y);
    *(unsigned*)(bs + (bn + 6) * LDA + bkp) = pack2bf(bA1.z, bB1.z);
    *(unsigned*)(bs + (bn + 7) * LDA + bkp) = pack2bf(bA1.w, bB1.w);
  };

  v8f c00 = {}, c01 = {}, c10 = {}, c11 = {};

  load_slab(0);
  store_slab(0);
  __syncthreads();

  const int NS = D_ / 32;
  for (int s = 0; s < NS; ++s) {
    const int cur = s & 1;
    if (s + 1 < NS) load_slab((s + 1) * 32);

    FragBF a0, a1, f0, f1;
    a0.q[0] = *(const uint4*)(&As[cur][(moff + lrow) * LDA + khalf * 8]);
    a0.q[1] = *(const uint4*)(&As[cur][(moff + lrow) * LDA + 16 + khalf * 8]);
    a1.q[0] = *(const uint4*)(&As[cur][(moff + 16 + lrow) * LDA + khalf * 8]);
    a1.q[1] = *(const uint4*)(&As[cur][(moff + 16 + lrow) * LDA + 16 + khalf * 8]);
    f0.q[0] = *(const uint4*)(&Bs[cur][(noff + lrow) * LDA + khalf * 8]);
    f0.q[1] = *(const uint4*)(&Bs[cur][(noff + lrow) * LDA + 16 + khalf * 8]);
    f1.q[0] = *(const uint4*)(&Bs[cur][(noff + 16 + lrow) * LDA + khalf * 8]);
    f1.q[1] = *(const uint4*)(&Bs[cur][(noff + 16 + lrow) * LDA + 16 + khalf * 8]);

    c00 = __builtin_amdgcn_wmma_f32_16x16x32_bf16(false, a0.v, false, f0.v,
                                                  (short)0, c00, false, false);
    c01 = __builtin_amdgcn_wmma_f32_16x16x32_bf16(false, a0.v, false, f1.v,
                                                  (short)0, c01, false, false);
    c10 = __builtin_amdgcn_wmma_f32_16x16x32_bf16(false, a1.v, false, f0.v,
                                                  (short)0, c10, false, false);
    c11 = __builtin_amdgcn_wmma_f32_16x16x32_bf16(false, a1.v, false, f1.v,
                                                  (short)0, c11, false, false);

    if (s + 1 < NS) {
      store_slab(cur ^ 1);
      __syncthreads();
    }
  }

  // epilogue: C layout -> M = r + 8*khalf, N = lane&15
  const int col0 = nt * 128 + noff + lrow;
  const int col1 = col0 + 16;
#pragma unroll
  for (int r = 0; r < 8; ++r) {
    int m0 = mt * 64 + moff + r + khalf * 8;
    int m1 = m0 + 16;
    if (m0 < cnt) {
      size_t hrow = (size_t)(base + m0) * H_;
      float v0 = c00[r] + b1[e * H_ + col0];
      v0 = 0.5f * v0 * (1.0f + erff(v0 * 0.70710678118654752f));  // exact GELU
      hbuf[hrow + col0] = f2bf(v0);
      float v1 = c01[r] + b1[e * H_ + col1];
      v1 = 0.5f * v1 * (1.0f + erff(v1 * 0.70710678118654752f));
      hbuf[hrow + col1] = f2bf(v1);
    }
    if (m1 < cnt) {
      size_t hrow = (size_t)(base + m1) * H_;
      float v0 = c10[r] + b1[e * H_ + col0];
      v0 = 0.5f * v0 * (1.0f + erff(v0 * 0.70710678118654752f));
      hbuf[hrow + col0] = f2bf(v0);
      float v1 = c11[r] + b1[e * H_ + col1];
      v1 = 0.5f * v1 * (1.0f + erff(v1 * 0.70710678118654752f));
      hbuf[hrow + col1] = f2bf(v1);
    }
  }
}

// ---------------------------------------------------------------------------
// GEMM2: y = (h[rows,H] @ w2[e] + b2[e]) * routing_weight -> fp32 [rows,H]
// Same tiling/pipeline as GEMM1; A rows are compact (no gather).
// ---------------------------------------------------------------------------
__global__ __launch_bounds__(256)
void gemm2_kernel(const unsigned short* __restrict__ hbuf,
                  const float* __restrict__ w2,
                  const float* __restrict__ b2,
                  const int* __restrict__ counts,
                  const int* __restrict__ offsets,
                  const float* __restrict__ wgt_list,
                  float* __restrict__ ybuf) {
  const int e = blockIdx.z, mt = blockIdx.y, nt = blockIdx.x;
  const int cnt = counts[e];
  if (mt * 64 >= cnt) return;
  const int base = offsets[e];

  __shared__ __align__(16) unsigned short As[2][64 * LDA];
  __shared__ __align__(16) unsigned short Bs[2][128 * LDA];

  const int tid = threadIdx.x;
  const int wave = tid >> 5, lane = tid & 31;
  const int moff = (wave & 1) * 32, noff = (wave >> 1) * 32;
  const int lrow = lane & 15, khalf = lane >> 4;

  const int arow = tid >> 2;
  const int aseg = (tid & 3) * 8;
  const int gra = mt * 64 + arow;
  const unsigned short* asrc =
      (gra < cnt) ? (hbuf + (size_t)(base + gra) * H_ + aseg) : nullptr;

  const int bn = (tid & 15) * 8;
  const int bkp = (tid >> 4) * 2;
  const float* wsrc = w2 + (size_t)e * H_ * H_ + (size_t)nt * 128 + bn;

  uint4 areg;
  float4 bA0, bA1, bB0, bB1;

  auto load_slab = [&](int k0) {
    areg = make_uint4(0u, 0u, 0u, 0u);
    if (asrc) areg = *(const uint4*)(asrc + k0);
    const float* p0 = wsrc + (size_t)(k0 + bkp) * H_;
    bA0 = *(const float4*)(p0);
    bA1 = *(const float4*)(p0 + 4);
    bB0 = *(const float4*)(p0 + H_);
    bB1 = *(const float4*)(p0 + H_ + 4);
  };
  auto store_slab = [&](int buf) {
    *(uint4*)(&As[buf][arow * LDA + aseg]) = areg;
    unsigned short* bs = &Bs[buf][0];
    *(unsigned*)(bs + (bn + 0) * LDA + bkp) = pack2bf(bA0.x, bB0.x);
    *(unsigned*)(bs + (bn + 1) * LDA + bkp) = pack2bf(bA0.y, bB0.y);
    *(unsigned*)(bs + (bn + 2) * LDA + bkp) = pack2bf(bA0.z, bB0.z);
    *(unsigned*)(bs + (bn + 3) * LDA + bkp) = pack2bf(bA0.w, bB0.w);
    *(unsigned*)(bs + (bn + 4) * LDA + bkp) = pack2bf(bA1.x, bB1.x);
    *(unsigned*)(bs + (bn + 5) * LDA + bkp) = pack2bf(bA1.y, bB1.y);
    *(unsigned*)(bs + (bn + 6) * LDA + bkp) = pack2bf(bA1.z, bB1.z);
    *(unsigned*)(bs + (bn + 7) * LDA + bkp) = pack2bf(bA1.w, bB1.w);
  };

  v8f c00 = {}, c01 = {}, c10 = {}, c11 = {};

  load_slab(0);
  store_slab(0);
  __syncthreads();

  const int NS = H_ / 32;
  for (int s = 0; s < NS; ++s) {
    const int cur = s & 1;
    if (s + 1 < NS) load_slab((s + 1) * 32);

    FragBF a0, a1, f0, f1;
    a0.q[0] = *(const uint4*)(&As[cur][(moff + lrow) * LDA + khalf * 8]);
    a0.q[1] = *(const uint4*)(&As[cur][(moff + lrow) * LDA + 16 + khalf * 8]);
    a1.q[0] = *(const uint4*)(&As[cur][(moff + 16 + lrow) * LDA + khalf * 8]);
    a1.q[1] = *(const uint4*)(&As[cur][(moff + 16 + lrow) * LDA + 16 + khalf * 8]);
    f0.q[0] = *(const uint4*)(&Bs[cur][(noff + lrow) * LDA + khalf * 8]);
    f0.q[1] = *(const uint4*)(&Bs[cur][(noff + lrow) * LDA + 16 + khalf * 8]);
    f1.q[0] = *(const uint4*)(&Bs[cur][(noff + 16 + lrow) * LDA + khalf * 8]);
    f1.q[1] = *(const uint4*)(&Bs[cur][(noff + 16 + lrow) * LDA + 16 + khalf * 8]);

    c00 = __builtin_amdgcn_wmma_f32_16x16x32_bf16(false, a0.v, false, f0.v,
                                                  (short)0, c00, false, false);
    c01 = __builtin_amdgcn_wmma_f32_16x16x32_bf16(false, a0.v, false, f1.v,
                                                  (short)0, c01, false, false);
    c10 = __builtin_amdgcn_wmma_f32_16x16x32_bf16(false, a1.v, false, f0.v,
                                                  (short)0, c10, false, false);
    c11 = __builtin_amdgcn_wmma_f32_16x16x32_bf16(false, a1.v, false, f1.v,
                                                  (short)0, c11, false, false);

    if (s + 1 < NS) {
      store_slab(cur ^ 1);
      __syncthreads();
    }
  }

  const int col0 = nt * 128 + noff + lrow;
  const int col1 = col0 + 16;
#pragma unroll
  for (int r = 0; r < 8; ++r) {
    int m0 = mt * 64 + moff + r + khalf * 8;
    int m1 = m0 + 16;
    if (m0 < cnt) {
      float wgt = wgt_list[e * BT_ + m0];
      size_t yrow = (size_t)(base + m0) * H_;
      ybuf[yrow + col0] = (c00[r] + b2[e * H_ + col0]) * wgt;
      ybuf[yrow + col1] = (c01[r] + b2[e * H_ + col1]) * wgt;
    }
    if (m1 < cnt) {
      float wgt = wgt_list[e * BT_ + m1];
      size_t yrow = (size_t)(base + m1) * H_;
      ybuf[yrow + col0] = (c10[r] + b2[e * H_ + col0]) * wgt;
      ybuf[yrow + col1] = (c11[r] + b2[e * H_ + col1]) * wgt;
    }
  }
}

// ---------------------------------------------------------------------------
// Combine: out[t,:] = y[row(t,0),:] + y[row(t,1),:]  (rows pre-weighted)
// ---------------------------------------------------------------------------
__global__ void combine_kernel(const float* __restrict__ ybuf,
                               const int* __restrict__ offsets,
                               const int* __restrict__ tok_e,
                               const int* __restrict__ tok_s,
                               float* __restrict__ out) {
  int idx = blockIdx.x * blockDim.x + threadIdx.x;
  int t = idx / (H_ / 4);
  int c = (idx % (H_ / 4)) * 4;
  int r0 = offsets[tok_e[2 * t + 0]] + tok_s[2 * t + 0];
  int r1 = offsets[tok_e[2 * t + 1]] + tok_s[2 * t + 1];
  float4 a = *(const float4*)(ybuf + (size_t)r0 * H_ + c);
  float4 b = *(const float4*)(ybuf + (size_t)r1 * H_ + c);
  float4 o;
  o.x = a.x + b.x; o.y = a.y + b.y; o.z = a.z + b.z; o.w = a.w + b.w;
  *(float4*)(out + (size_t)t * H_ + c) = o;
}

// ---------------------------------------------------------------------------
extern "C" void kernel_launch(void* const* d_in, const int* in_sizes, int n_in,
                              void* d_out, int out_size, void* d_ws,
                              size_t ws_size, hipStream_t stream) {
  (void)in_sizes; (void)n_in; (void)out_size; (void)ws_size;
  const float* x  = (const float*)d_in[0];  // [B,T,D]
  const float* gw = (const float*)d_in[1];  // [D,E]
  const float* w1 = (const float*)d_in[2];  // [E,D,H]
  const float* b1 = (const float*)d_in[3];  // [E,H]
  const float* w2 = (const float*)d_in[4];  // [E,H,H]
  const float* b2 = (const float*)d_in[5];  // [E,H]
  float* out = (float*)d_out;               // [B,T,H] fp32

  // workspace carve (256B aligned), total ~59 MB
  char* ws = (char*)d_ws;
  size_t off = 0;
  auto carve = [&](size_t bytes) -> char* {
    char* p = ws + off;
    off = (off + bytes + 255) & ~(size_t)255;
    return p;
  };
  int*            counts   = (int*)carve(E_ * sizeof(int));
  int*            offsets  = (int*)carve((E_ + 1) * sizeof(int));
  int*            tok_list = (int*)carve((size_t)E_ * BT_ * sizeof(int));
  float*          wgt_list = (float*)carve((size_t)E_ * BT_ * sizeof(float));
  int*            tok_e    = (int*)carve((size_t)BT_ * TOPK_ * sizeof(int));
  int*            tok_s    = (int*)carve((size_t)BT_ * TOPK_ * sizeof(int));
  unsigned short* xb       = (unsigned short*)carve((size_t)BT_ * D_ * 2);
  unsigned short* hbuf     = (unsigned short*)carve((size_t)BT_ * TOPK_ * H_ * 2);
  float*          ybuf     = (float*)carve((size_t)BT_ * TOPK_ * H_ * 4);

  hipMemsetAsync(counts, 0, E_ * sizeof(int), stream);

  cvt_x_kernel<<<(BT_ * D_) / 1024, 256, 0, stream>>>(x, (unsigned*)xb);
  gate_kernel<<<BT_ / 8, 256, 0, stream>>>(x, gw, counts, tok_list, wgt_list,
                                           tok_e, tok_s);
  scan_kernel<<<1, 32, 0, stream>>>(counts, offsets);

  // expert-major grid: all tiles of one expert scheduled together so the
  // expert's weights (<= 67 MB) stay resident in the 192 MB L2
  dim3 gg(H_ / 128, (BT_ + 63) / 64, E_);
  gemm1_kernel<<<gg, 256, 0, stream>>>(xb, w1, b1, counts, offsets, tok_list,
                                       hbuf);
  gemm2_kernel<<<gg, 256, 0, stream>>>(hbuf, w2, b2, counts, offsets, wgt_list,
                                       ybuf);
  combine_kernel<<<(BT_ * H_ / 4) / 256, 256, 0, stream>>>(ybuf, offsets,
                                                           tok_e, tok_s, out);
}
